// RootCentricTreeLSTM_83099027243631
// MI455X (gfx1250) — matrix operand order repaired and checked by reading, not compiled, not run on hardware
//
#include <hip/hip_runtime.h>
#include <hip/hip_bf16.h>

// ---------------------------------------------------------------------------
// Root-centric TreeLSTM on gfx1250 (MI455X).
//  * bf16 WMMA (v_wmma_f32_16x16x32_bf16), f32 accumulation, 5 gate accums
//  * A panels: GLOBAL_LOAD_ASYNC_TO_LDS_B128 into double-buffered LDS,
//    overlapped with compute via ASYNCcnt (s_wait_asynccnt) + barriers
//  * B (weights) from global, L2/WGP$-resident, 1 tile feeds 2 WMMAs
//  * block = 8 waves: 32 rows x 128 h-cols; wave = 2 Msubtiles x 16 cols x 5 gates
// ---------------------------------------------------------------------------

typedef __attribute__((ext_vector_type(16))) __bf16 v16bf;
typedef __attribute__((ext_vector_type(8)))  __bf16 v8bf;
typedef __attribute__((ext_vector_type(8)))  float  v8f;
typedef __attribute__((ext_vector_type(4)))  float  v4f;

#define B_  32
#define NN  4095        // 2^12 - 1 nodes
#define F_  256
#define H_  256
#define G5  1280        // 5*H
#define RS  264         // LDS A-panel row stride in bf16 elems (528B -> 2-way conflicts max)

// fast sigmoid / tanh via v_exp_f32 (TRANS op, co-executes with WMMA)
__device__ __forceinline__ float sigm(float x) {
    return 1.0f / (1.0f + __expf(-x));
}
__device__ __forceinline__ float ftanh(float x) {
    const float e = __expf(2.0f * x);
    return (e - 1.0f) / (e + 1.0f);
}

// ---- A operand from LDS panel row: 16x32 bf16 tile -------------------------
// lane<16 : M=lane,    K in {k0+0..7}  U {k0+16..23}
// lane>=16: M=lane-16, K in {k0+8..15} U {k0+24..31}
__device__ __forceinline__ v16bf load_a_lds(const __bf16* rowp, int koff) {
    v8bf lo = *(const v8bf*)(rowp + koff);        // ds_load_b128
    v8bf hi = *(const v8bf*)(rowp + koff + 16);   // ds_load_b128
    v16bf a;
#pragma unroll
    for (int i = 0; i < 8; ++i) { a[i] = lo[i]; a[8 + i] = hi[i]; }
    return a;
}

// ---- B operand: 32x16 bf16 tile, weight row (=output column) per lane ------
__device__ __forceinline__ v16bf load_b_w(const __bf16* __restrict__ wrow,
                                          int k0, int lane) {
    const int kb = k0 + ((lane & 16) ? 8 : 0);
    const v8bf* p = (const v8bf*)(wrow + kb);     // 2x global_load_b128
    v8bf lo = p[0], hi = p[2];
    v16bf b;
#pragma unroll
    for (int i = 0; i < 8; ++i) { b[i] = lo[i]; b[8 + i] = hi[i]; }
    return b;
}

// ---------------------------------------------------------------------------
// One tree level. n = 2^lgn nodes at index s.. ; row m = b*n + node_local.
// Grid: 2*n blocks x 256 threads. mb = blk>>1 (32 rows), hb = blk&1 (128 cols).
// ---------------------------------------------------------------------------
template <bool LEAF>
__global__ __launch_bounds__(256) void tree_level_kernel(
    const __bf16* __restrict__ feat_bf,  // (B, NN, F) bf16
    const __bf16* __restrict__ Wbf,      // [Wx | Wl | Wr], each (G5, F) row-major bf16
    const float*  __restrict__ bx,
    const float*  __restrict__ bl,
    const float*  __restrict__ br,
    __bf16* __restrict__ h_bf,           // (B, NN, H) bf16 (GEMM operand for parents)
    float*  __restrict__ c_all,          // (B, NN, H) f32
    float*  __restrict__ out,            // non-null only at root level: (B, H) f32
    int s, int lgn) {
    const int n    = 1 << lgn;
    const int lane = threadIdx.x & 31;
    const int w    = threadIdx.x >> 5;         // wave in block: 0..7
    const int mb   = blockIdx.x >> 1;
    const int hb   = blockIdx.x & 1;
    const int m0   = mb << 5;                  // 32 rows per block
    const int col  = hb * 128 + w * 16 + (lane & 15);

    __shared__ __bf16 apanel[2][32 * RS];      // 2 x 16.5 KB A panels (double buffer)

    // ---- per-thread staging chunk geometry (computed once) -----------------
    // 1024 16B chunks per panel; this thread owns 4 of them.
    size_t  st_rowbase[4];   // (size_t)b * NN
    int     st_nd[4];        // global node index of the chunk's row
    int     st_eo[4];        // element offset within the 256-elem row
    int     st_lrow[4];      // LDS row
#pragma unroll
    for (int i = 0; i < 4; ++i) {
        const int c   = (int)threadIdx.x + (i << 8);
        const int row = c >> 5;
        const int m   = m0 + row;
        const int b   = m >> lgn;
        st_lrow[i]    = row;
        st_eo[i]      = (c & 31) << 3;
        st_rowbase[i] = (size_t)b * NN;
        st_nd[i]      = s + (m & (n - 1));
    }

    // issue one panel's async loads (no wait) --------------------------------
    // cs: 0 -> node itself (features), 1 -> left child h, 2 -> right child h
    auto stage_issue = [&](int buf, const __bf16* __restrict__ mat, int cs) {
#pragma unroll
        for (int i = 0; i < 4; ++i) {
            const int src = (cs == 0) ? st_nd[i] : (2 * st_nd[i] + cs);
            const __bf16* gp = mat + (st_rowbase[i] + src) * 256 + st_eo[i];
            const uint32_t la = (uint32_t)(uintptr_t)(const void*)
                                (apanel[buf] + st_lrow[i] * RS + st_eo[i]);
            asm volatile("global_load_async_to_lds_b128 %0, %1, off"
                         :: "v"(la), "v"((uint64_t)(uintptr_t)gp) : "memory");
        }
    };
    auto stage_sync = [&]() {
        asm volatile("s_wait_asynccnt 0x0" ::: "memory");
        __syncthreads();
    };

    v8f acc0[5], acc1[5];
    v8f z = {};
#pragma unroll
    for (int g = 0; g < 5; ++g) { acc0[g] = z; acc1[g] = z; }

    // ---- 8 K-steps over one staged panel against one weight region ---------
    auto do_region = [&](int buf, const __bf16* __restrict__ wreg) {
        const __bf16* r0 = apanel[buf] + (lane & 15) * RS;         // Msub 0
        const __bf16* r1 = apanel[buf] + (16 + (lane & 15)) * RS;  // Msub 1
#pragma unroll
        for (int step = 0; step < 8; ++step) {
            const int k0   = step << 5;
            const int koff = k0 + ((lane & 16) ? 8 : 0);
            v16bf a0 = load_a_lds(r0, koff);
            v16bf a1 = load_a_lds(r1, koff);
#pragma unroll
            for (int g = 0; g < 5; ++g) {
                const __bf16* wrow = wreg + ((size_t)(g * H_ + col)) * F_;
                v16bf bv = load_b_w(wrow, k0, lane);
                acc0[g] = __builtin_amdgcn_wmma_f32_16x16x32_bf16(
                    false, a0, false, bv, (short)0, acc0[g], false, false);
                acc1[g] = __builtin_amdgcn_wmma_f32_16x16x32_bf16(
                    false, a1, false, bv, (short)0, acc1[g], false, false);
            }
        }
    };

    // ---- software pipeline: stage region r+1 while computing region r ------
    stage_issue(0, feat_bf, 0);
    stage_sync();
    if (!LEAF) stage_issue(1, h_bf, 1);          // prefetch left-child panel
    do_region(0, Wbf);                           // features x Wx
    if (!LEAF) {
        stage_sync();                            // buf1 ready; buf0 free
        stage_issue(0, h_bf, 2);                 // prefetch right-child panel
        do_region(1, Wbf + (size_t)1 * G5 * F_); // left h x Wl
        stage_sync();
        do_region(0, Wbf + (size_t)2 * G5 * F_); // right h x Wr
    }

    // ---- biases for this column (leaves use bx only, per reference) --------
    float bias[5];
#pragma unroll
    for (int g = 0; g < 5; ++g) {
        const int gc = g * H_ + col;
        bias[g] = bx[gc];
        if (!LEAF) bias[g] += bl[gc] + br[gc];
    }

    // ---- gates epilogue; C layout: VGPR j -> row +j (lanes<16) / +j+8 ------
#pragma unroll
    for (int ms = 0; ms < 2; ++ms) {
        v8f* acc = ms ? acc1 : acc0;
#pragma unroll
        for (int j = 0; j < 8; ++j) {
            const int row  = m0 + ms * 16 + j + ((lane & 16) ? 8 : 0);
            const int b_r  = row >> lgn;
            const int nd_r = s + (row & (n - 1));
            const size_t base = ((size_t)b_r * NN + nd_r) * H_ + col;

            const float iv = sigm(acc[0][j] + bias[0]);
            const float ov = sigm(acc[1][j] + bias[1]);
            const float fl = sigm(acc[2][j] + bias[2]);
            const float fr = sigm(acc[3][j] + bias[3]);
            const float uv = ftanh(acc[4][j] + bias[4]);

            float lc = 0.0f, rc = 0.0f;
            if (!LEAF) {
                lc = c_all[((size_t)b_r * NN + (2 * nd_r + 1)) * H_ + col];
                rc = c_all[((size_t)b_r * NN + (2 * nd_r + 2)) * H_ + col];
            }
            const float cv = iv * uv + fl * lc + fr * rc;
            const float hv = ov * ftanh(cv);
            c_all[base] = cv;
            h_bf[base]  = (__bf16)hv;
            if (out) out[(size_t)b_r * H_ + col] = hv;   // root: n==1, nd_r==0
        }
    }
}

// ---- fp32 -> bf16 streaming conversion (8 elems/thread) --------------------
__global__ __launch_bounds__(256) void cvt_bf16_kernel(
    const float* __restrict__ src, __bf16* __restrict__ dst, int nelem) {
    const int i = (blockIdx.x * blockDim.x + threadIdx.x) << 3;
    if (i < nelem) {
        v4f a = *(const v4f*)(src + i);
        v4f b = *(const v4f*)(src + i + 4);
        v8bf o;
#pragma unroll
        for (int k = 0; k < 4; ++k) { o[k] = (__bf16)a[k]; o[4 + k] = (__bf16)b[k]; }
        *(v8bf*)(dst + i) = o;
    }
}

// ---------------------------------------------------------------------------
extern "C" void kernel_launch(void* const* d_in, const int* in_sizes, int n_in,
                              void* d_out, int out_size, void* d_ws, size_t ws_size,
                              hipStream_t stream) {
    (void)in_sizes; (void)n_in; (void)out_size; (void)ws_size;
    const float* feat = (const float*)d_in[0];   // (B, NN, F)
    const float* Wx   = (const float*)d_in[1];   // (G5, F)
    const float* Wl   = (const float*)d_in[2];   // (G5, H)
    const float* Wr   = (const float*)d_in[3];   // (G5, H)
    const float* bx   = (const float*)d_in[4];
    const float* bl   = (const float*)d_in[5];
    const float* br   = (const float*)d_in[6];
    float* out = (float*)d_out;

    // ws: Wbf (1.9MB) | feat_bf (67MB) | h_bf (67MB) | c_all (134MB)
    char* ws = (char*)d_ws;
    __bf16* Wbf = (__bf16*)ws;
    size_t off = (size_t)3 * G5 * F_ * sizeof(__bf16);
    __bf16* feat_bf = (__bf16*)(ws + off);
    off += (size_t)B_ * NN * F_ * sizeof(__bf16);
    __bf16* h_bf = (__bf16*)(ws + off);
    off += (size_t)B_ * NN * H_ * sizeof(__bf16);
    float* c_all = (float*)(ws + off);

    // 1) bf16 conversions (weights ~1M elems, features ~33.5M elems)
    {
        const int welem = G5 * F_;                      // 327,680
        cvt_bf16_kernel<<<(welem / 8 + 255) / 256, 256, 0, stream>>>(Wx, Wbf, welem);
        cvt_bf16_kernel<<<(welem / 8 + 255) / 256, 256, 0, stream>>>(Wl, Wbf + welem, welem);
        cvt_bf16_kernel<<<(welem / 8 + 255) / 256, 256, 0, stream>>>(Wr, Wbf + 2 * welem, welem);
        const int felem = B_ * NN * F_;                 // 33,546,240
        cvt_bf16_kernel<<<(felem / 8 + 255) / 256, 256, 0, stream>>>(feat, feat_bf, felem);
    }

    // 2) leaf level: lvl=11, s=2047, n=2048 (features only; c_l = c_r = 0)
    tree_level_kernel<true><<<2 * 2048, 256, 0, stream>>>(
        feat_bf, Wbf, bx, bl, br, h_bf, c_all, nullptr, 2047, 11);

    // 3) internal levels bottom-up; level 0 writes root h straight to out
    for (int lvl = 10; lvl >= 0; --lvl) {
        const int n = 1 << lvl;
        tree_level_kernel<false><<<2 * n, 256, 0, stream>>>(
            feat_bf, Wbf, bx, bl, br, h_bf, c_all,
            (lvl == 0) ? out : nullptr, n - 1, lvl);
    }
}
